// MPNNLayer_57123065037603
// MI455X (gfx1250) — matrix operand order repaired and checked by reading, not compiled, not run on hardware
//
#include <hip/hip_runtime.h>
#include <hip/hip_bf16.h>

#define NN 10000
#define EE 160000
#define HH 512
#define EDD 16
#define LN_EPS 1e-5f
#define ESCALE 0.1f

typedef __bf16 v16bf __attribute__((ext_vector_type(16)));
typedef float  v8f   __attribute__((ext_vector_type(8)));

union BF { v16bf v; unsigned int u[8]; };

__device__ __forceinline__ unsigned short f2bf(float f) {
    unsigned int u = __float_as_uint(f);
    return (unsigned short)((u + 0x7FFFu + ((u >> 16) & 1u)) >> 16);
}
__device__ __forceinline__ float bf2f(unsigned short h) {
    return __uint_as_float(((unsigned int)h) << 16);
}
__device__ __forceinline__ float gelu_f(float x) {
    return 0.5f * x * (1.0f + erff(x * 0.70710678118654752440f));
}

// A fragment (16xK tile, row-major bf16 in LDS), per CDNA5 16-bit A layout:
// lanes 0-15: row=lane, K 0..7 in v0..3, K 16..23 in v4..7
// lanes 16-31: row=lane-16, K 8..15 in v0..3, K 24..31 in v4..7
__device__ __forceinline__ v16bf ldsA(const unsigned short* z, int stride, int k0) {
    int lane = threadIdx.x & 31;
    int row = lane & 15;
    int kb = (lane & 16) ? 8 : 0;
    const unsigned short* p = z + row * stride + k0 + kb;
    BF f;
#pragma unroll
    for (int i = 0; i < 4; ++i) f.u[i]     = *(const unsigned int*)(p + 2 * i);
#pragma unroll
    for (int i = 0; i < 4; ++i) f.u[4 + i] = *(const unsigned int*)(p + 16 + 2 * i);
    return f.v;
}

// B fragment from pre-swizzled weights: tile (tk,tn) stored as 32 lanes x 8 dwords,
// so each lane issues two contiguous global_load_b128.
__device__ __forceinline__ v16bf glbB(const unsigned int* __restrict__ w, int ntn, int tk, int tn) {
    int lane = threadIdx.x & 31;
    const unsigned int* p = w + (((size_t)tk * ntn + tn) * 32 + lane) * 8;
    BF f;
#pragma unroll
    for (int j = 0; j < 8; ++j) f.u[j] = p[j];
    return f.v;
}

__device__ __forceinline__ v8f wmma_bf16(v16bf a, v16bf b, v8f c) {
    return __builtin_amdgcn_wmma_f32_16x16x32_bf16(false, a, false, b, (short)0, c, false, false);
}

// Software-pipelined 16xN GEMM: 4 N-tiles per wave, double-buffered A/B fragments
// so the k+1 loads are in flight while k's WMMAs execute.
template<int NT>
__device__ __forceinline__ void gemm4(const unsigned short* aLds, int stride,
                                      const unsigned int* __restrict__ w, int ntn,
                                      int tnb, v8f acc[4]) {
    v16bf a0 = ldsA(aLds, stride, 0);
    v16bf b0[4];
#pragma unroll
    for (int i = 0; i < 4; ++i) b0[i] = glbB(w, ntn, 0, tnb + i);
#pragma unroll 1
    for (int tk = 0; tk < NT - 1; ++tk) {
        v16bf a1 = ldsA(aLds, stride, (tk + 1) * 32);
        v16bf b1[4];
#pragma unroll
        for (int i = 0; i < 4; ++i) b1[i] = glbB(w, ntn, tk + 1, tnb + i);
#pragma unroll
        for (int i = 0; i < 4; ++i) acc[i] = wmma_bf16(a0, b0[i], acc[i]);
        a0 = a1;
#pragma unroll
        for (int i = 0; i < 4; ++i) b0[i] = b1[i];
    }
#pragma unroll
    for (int i = 0; i < 4; ++i) acc[i] = wmma_bf16(a0, b0[i], acc[i]);
}

// ---------------- prep kernels ----------------

__global__ void zero_kernel(float* __restrict__ p, int n) {
    for (int i = blockIdx.x * blockDim.x + threadIdx.x; i < n; i += gridDim.x * blockDim.x)
        p[i] = 0.0f;
}

// Convert f32 weight (realK x N row-major) to bf16, zero-padded to Kpad,
// swizzled to WMMA B-fragment tiles.
__global__ void wswz_kernel(const float* __restrict__ src, unsigned int* __restrict__ dst,
                            int Kpad, int N, int realK) {
    int ntn = N >> 4;
    int total = (Kpad >> 5) * ntn * 256;
    for (int i = blockIdx.x * blockDim.x + threadIdx.x; i < total; i += gridDim.x * blockDim.x) {
        int j = i & 7;
        int lane = (i >> 3) & 31;
        int tile = i >> 8;
        int tn = tile % ntn;
        int tk = tile / ntn;
        int col = tn * 16 + (lane & 15);
        int kb = (lane & 16) ? 8 : 0;
        int k = tk * 32 + kb + 2 * (j & 3) + ((j & 4) ? 16 : 0);
        float lo = (k     < realK) ? src[(size_t)k * N + col]       : 0.0f;
        float hi = (k + 1 < realK) ? src[(size_t)(k + 1) * N + col] : 0.0f;
        dst[i] = (unsigned int)f2bf(lo) | ((unsigned int)f2bf(hi) << 16);
    }
}

// ---------------- edge kernel ----------------

struct EdgeSmem {
    unsigned short z[16 * 1056];   // [hs(512) | hd(512)->e_new(16)+pad | ea(16) | pad(16)]
    unsigned short m[16 * 512];    // gelu(z @ msg_w1 + b1) in bf16
    unsigned short t[16 * 32];     // gelu(z @ em_w1 + b1), K-padded to 32
    float delta[16 * 16];
    float eaf[16 * 16];
    float gate[16];
    int srcl[16];
    int dstl[16];
};

__global__ __launch_bounds__(256)
void edge_kernel(const float* __restrict__ h,
                 const float* __restrict__ edge_attr,
                 const int* __restrict__ edge_index,
                 const unsigned int* __restrict__ msg_w1sw,
                 const float* __restrict__ msg_b1,
                 const unsigned int* __restrict__ msg_w2sw,
                 const float* __restrict__ msg_b2,
                 const unsigned int* __restrict__ em_w1sw,
                 const float* __restrict__ em_b1,
                 const unsigned int* __restrict__ em_w2sw,
                 const float* __restrict__ em_b2,
                 const float* __restrict__ eg_w,
                 const float* __restrict__ eg_b,
                 const float* __restrict__ en_g,
                 const float* __restrict__ en_b,
                 float* __restrict__ agg,
                 float* __restrict__ e_out) {
    __shared__ EdgeSmem s;
    const int e0 = blockIdx.x * 16;
    const int tid = threadIdx.x;
    const int lane = tid & 31;
    const int wid = tid >> 5;

    if (tid < 16) {
        s.srcl[tid] = edge_index[e0 + tid];
        s.dstl[tid] = edge_index[EE + e0 + tid];
    }
    __syncthreads();

    // Phase 1: gather hs/hd (L2-resident h) with b128 loads, pack to bf16 pairs.
    for (int idx = tid; idx < 16 * 128; idx += 256) {
        int row = idx >> 7, c4 = idx & 127;
        const float4* hrs = (const float4*)(h + (size_t)s.srcl[row] * HH);
        const float4* hrd = (const float4*)(h + (size_t)s.dstl[row] * HH);
        float4 vs = hrs[c4];
        float4 vd = hrd[c4];
        uint2 ps, pd;
        ps.x = (unsigned int)f2bf(vs.x) | ((unsigned int)f2bf(vs.y) << 16);
        ps.y = (unsigned int)f2bf(vs.z) | ((unsigned int)f2bf(vs.w) << 16);
        pd.x = (unsigned int)f2bf(vd.x) | ((unsigned int)f2bf(vd.y) << 16);
        pd.y = (unsigned int)f2bf(vd.z) | ((unsigned int)f2bf(vd.w) << 16);
        *(uint2*)&s.z[row * 1056 + c4 * 4]       = ps;
        *(uint2*)&s.z[row * 1056 + 512 + c4 * 4] = pd;
    }
    {
        int row = tid >> 4, col = tid & 15;
        float ea = edge_attr[(size_t)(e0 + row) * EDD + col];
        s.eaf[row * 16 + col] = ea;
        s.z[row * 1056 + 1024 + col] = f2bf(ea);
        s.z[row * 1056 + 1040 + col] = 0;   // K pad to 1056
    }
    __syncthreads();

    // Phase 2: wave0: edge-MLP first GEMM (16x16, K=1056, pipelined).
    //          wave1: gate dot-product.
    if (wid == 0) {
        v8f acc = {};
        v16bf a0 = ldsA(s.z, 1056, 0);
        v16bf b0 = glbB(em_w1sw, 1, 0, 0);
#pragma unroll 1
        for (int tk = 0; tk < 32; ++tk) {
            v16bf a1 = ldsA(s.z, 1056, (tk + 1) * 32);
            v16bf b1 = glbB(em_w1sw, 1, tk + 1, 0);
            acc = wmma_bf16(a0, b0, acc);
            a0 = a1; b0 = b1;
        }
        acc = wmma_bf16(a0, b0, acc);
        int col = lane & 15;
        int rb = (lane & 16) ? 8 : 0;
        float bias = em_b1[col];
#pragma unroll
        for (int r = 0; r < 8; ++r) {
            s.t[(rb + r) * 32 + col]      = f2bf(gelu_f(acc[r] + bias));
            s.t[(rb + r) * 32 + 16 + col] = 0;   // K pad to 32
        }
    } else if (wid == 1) {
        if (lane < 16) {
            float sum = eg_b[0];
            const unsigned short* zr = s.z + lane * 1056;
            for (int k = 0; k < 1040; ++k) sum += bf2f(zr[k]) * eg_w[k];
            s.gate[lane] = 1.0f / (1.0f + __expf(-sum));
        }
    }
    __syncthreads();

    // Phase 3: delta = t @ em_w2 + b2  (single WMMA, K padded to 32)
    if (wid == 0) {
        v8f acc = {};
        v16bf a = ldsA(s.t, 32, 0);
        v16bf b = glbB(em_w2sw, 1, 0, 0);
        acc = wmma_bf16(a, b, acc);
        int col = lane & 15;
        int rb = (lane & 16) ? 8 : 0;
        float bias = em_b2[col];
#pragma unroll
        for (int r = 0; r < 8; ++r) s.delta[(rb + r) * 16 + col] = acc[r] + bias;
    }
    __syncthreads();

    // Phase 4: e_new = LN(ea + 0.1*gate*delta); write out + splice into z tile.
    if (wid == 0 && lane < 16) {
        float v[16];
        float g = s.gate[lane];
        float mu = 0.f;
#pragma unroll
        for (int j = 0; j < 16; ++j) {
            v[j] = s.eaf[lane * 16 + j] + ESCALE * g * s.delta[lane * 16 + j];
            mu += v[j];
        }
        mu *= (1.0f / 16.0f);
        float var = 0.f;
#pragma unroll
        for (int j = 0; j < 16; ++j) { float d = v[j] - mu; var += d * d; }
        var *= (1.0f / 16.0f);
        float rs = rsqrtf(var + LN_EPS);
#pragma unroll
        for (int j = 0; j < 16; ++j) {
            float o = (v[j] - mu) * rs * en_g[j] + en_b[j];
            e_out[(size_t)(e0 + lane) * EDD + j] = o;
            s.z[lane * 1056 + 512 + j] = f2bf(o);   // zm = [hs | e_new]
            s.z[lane * 1056 + 528 + j] = 0;         // K pad to 544
        }
    }
    __syncthreads();

    // Phase 5: m = gelu(zm @ msg_w1 + b1), K=544, 4 N-tiles/wave, pipelined.
    {
        v8f acc[4] = {};
        const int tnb = wid * 4;
        gemm4<17>(s.z, 1056, msg_w1sw, 32, tnb, acc);
        int col0 = lane & 15;
        int rb = (lane & 16) ? 8 : 0;
#pragma unroll
        for (int i = 0; i < 4; ++i) {
            int col = (tnb + i) * 16 + col0;
            float bias = msg_b1[col];
#pragma unroll
            for (int r = 0; r < 8; ++r)
                s.m[(rb + r) * 512 + col] = f2bf(gelu_f(acc[i][r] + bias));
        }
    }
    __syncthreads();

    // Phase 6: msg = m @ msg_w2 + b2, then scatter-add into L2-resident agg.
    {
        v8f acc[4] = {};
        const int tnb = wid * 4;
        gemm4<16>(s.m, 512, msg_w2sw, 32, tnb, acc);
        int col0 = lane & 15;
        int rb = (lane & 16) ? 8 : 0;
#pragma unroll
        for (int i = 0; i < 4; ++i) {
            int col = (tnb + i) * 16 + col0;
            float bias = msg_b2[col];
#pragma unroll
            for (int r = 0; r < 8; ++r) {
                int d = s.dstl[rb + r];
                unsafeAtomicAdd(&agg[(size_t)d * HH + col], acc[i][r] + bias);
            }
        }
    }
}

// ---------------- node kernel ----------------

struct NodeSmem {
    union {
        unsigned short zu[16 * 1024];  // [h | agg] bf16
        float res[16 * 512];           // h2 + h (reused after GEMM1)
    } u;
    unsigned short m[16 * 512];
};

__global__ __launch_bounds__(256)
void node_kernel(const float* __restrict__ h,
                 const float* __restrict__ agg,
                 const unsigned int* __restrict__ upd_w1sw,
                 const float* __restrict__ upd_b1,
                 const unsigned int* __restrict__ upd_w2sw,
                 const float* __restrict__ upd_b2,
                 const float* __restrict__ norm_g,
                 const float* __restrict__ norm_b,
                 float* __restrict__ h_out) {
    __shared__ NodeSmem s;
    const int n0 = blockIdx.x * 16;
    const int tid = threadIdx.x;
    const int lane = tid & 31;
    const int wid = tid >> 5;

    for (int idx = tid; idx < 16 * 128; idx += 256) {
        int row = idx >> 7, c4 = idx & 127;
        const float4* hr = (const float4*)(h   + (size_t)(n0 + row) * HH);
        const float4* ar = (const float4*)(agg + (size_t)(n0 + row) * HH);
        float4 vh = hr[c4];
        float4 va = ar[c4];
        uint2 ph, pa;
        ph.x = (unsigned int)f2bf(vh.x) | ((unsigned int)f2bf(vh.y) << 16);
        ph.y = (unsigned int)f2bf(vh.z) | ((unsigned int)f2bf(vh.w) << 16);
        pa.x = (unsigned int)f2bf(va.x) | ((unsigned int)f2bf(va.y) << 16);
        pa.y = (unsigned int)f2bf(va.z) | ((unsigned int)f2bf(va.w) << 16);
        *(uint2*)&s.u.zu[row * 1024 + c4 * 4]       = ph;
        *(uint2*)&s.u.zu[row * 1024 + 512 + c4 * 4] = pa;
    }
    __syncthreads();

    // GEMM1: gelu([h|agg] @ upd_w1 + b1), K=1024, pipelined.
    {
        v8f acc[4] = {};
        const int tnb = wid * 4;
        gemm4<32>(s.u.zu, 1024, upd_w1sw, 32, tnb, acc);
        int col0 = lane & 15;
        int rb = (lane & 16) ? 8 : 0;
#pragma unroll
        for (int i = 0; i < 4; ++i) {
            int col = (tnb + i) * 16 + col0;
            float bias = upd_b1[col];
#pragma unroll
            for (int r = 0; r < 8; ++r)
                s.m[(rb + r) * 512 + col] = f2bf(gelu_f(acc[i][r] + bias));
        }
    }
    __syncthreads();

    // GEMM2 + residual: res = m @ upd_w2 + b2 + h
    {
        v8f acc[4] = {};
        const int tnb = wid * 4;
        gemm4<16>(s.m, 512, upd_w2sw, 32, tnb, acc);
        int col0 = lane & 15;
        int rb = (lane & 16) ? 8 : 0;
#pragma unroll
        for (int i = 0; i < 4; ++i) {
            int col = (tnb + i) * 16 + col0;
            float bias = upd_b2[col];
#pragma unroll
            for (int r = 0; r < 8; ++r) {
                int row = rb + r;
                s.u.res[row * 512 + col] =
                    acc[i][r] + bias + h[(size_t)(n0 + row) * HH + col];
            }
        }
    }
    __syncthreads();

    // LayerNorm over 512 per row; wave w handles rows 2w, 2w+1.
    for (int rr = 0; rr < 2; ++rr) {
        int row = wid * 2 + rr;
        float sum = 0.f, sq = 0.f;
#pragma unroll
        for (int j = 0; j < 16; ++j) {
            float v = s.u.res[row * 512 + lane + j * 32];
            sum += v; sq += v * v;
        }
#pragma unroll
        for (int off = 16; off > 0; off >>= 1) {
            sum += __shfl_xor(sum, off, 32);
            sq  += __shfl_xor(sq,  off, 32);
        }
        float mu = sum * (1.0f / 512.0f);
        float var = sq * (1.0f / 512.0f) - mu * mu;
        float rs = rsqrtf(var + LN_EPS);
#pragma unroll
        for (int j = 0; j < 16; ++j) {
            int col = lane + j * 32;
            float v = s.u.res[row * 512 + col];
            h_out[(size_t)(n0 + row) * HH + col] = (v - mu) * rs * norm_g[col] + norm_b[col];
        }
    }
}

// ---------------- launcher ----------------

extern "C" void kernel_launch(void* const* d_in, const int* in_sizes, int n_in,
                              void* d_out, int out_size, void* d_ws, size_t ws_size,
                              hipStream_t stream) {
    const float* h         = (const float*)d_in[0];
    const float* edge_attr = (const float*)d_in[1];
    const float* msg_w1 = (const float*)d_in[2];
    const float* msg_b1 = (const float*)d_in[3];
    const float* msg_w2 = (const float*)d_in[4];
    const float* msg_b2 = (const float*)d_in[5];
    const float* upd_w1 = (const float*)d_in[6];
    const float* upd_b1 = (const float*)d_in[7];
    const float* upd_w2 = (const float*)d_in[8];
    const float* upd_b2 = (const float*)d_in[9];
    const float* norm_g = (const float*)d_in[10];
    const float* norm_b = (const float*)d_in[11];
    const float* em_w1  = (const float*)d_in[12];
    const float* em_b1  = (const float*)d_in[13];
    const float* em_w2  = (const float*)d_in[14];
    const float* em_b2  = (const float*)d_in[15];
    const float* eg_w   = (const float*)d_in[16];
    const float* eg_b   = (const float*)d_in[17];
    const float* en_g   = (const float*)d_in[18];
    const float* en_b   = (const float*)d_in[19];
    const int*   edge_index = (const int*)d_in[20];

    char* ws = (char*)d_ws;
    auto alignup = [](size_t x) { return (x + 255) & ~(size_t)255; };
    size_t off = 0;
    float* agg = (float*)(ws + off);            off = alignup(off + (size_t)NN * HH * 4);
    unsigned int* msg_w1sw = (unsigned int*)(ws + off); off = alignup(off + (size_t)17 * 32 * 256 * 4);
    unsigned int* msg_w2sw = (unsigned int*)(ws + off); off = alignup(off + (size_t)16 * 32 * 256 * 4);
    unsigned int* upd_w1sw = (unsigned int*)(ws + off); off = alignup(off + (size_t)32 * 32 * 256 * 4);
    unsigned int* upd_w2sw = (unsigned int*)(ws + off); off = alignup(off + (size_t)16 * 32 * 256 * 4);
    unsigned int* em_w1sw  = (unsigned int*)(ws + off); off = alignup(off + (size_t)33 * 256 * 4);
    unsigned int* em_w2sw  = (unsigned int*)(ws + off); off = alignup(off + (size_t)256 * 4);

    zero_kernel<<<1024, 256, 0, stream>>>(agg, NN * HH);
    wswz_kernel<<<(17 * 32 * 256 + 255) / 256, 256, 0, stream>>>(msg_w1, msg_w1sw, 544, 512, 528);
    wswz_kernel<<<(16 * 32 * 256 + 255) / 256, 256, 0, stream>>>(msg_w2, msg_w2sw, 512, 512, 512);
    wswz_kernel<<<(32 * 32 * 256 + 255) / 256, 256, 0, stream>>>(upd_w1, upd_w1sw, 1024, 512, 1024);
    wswz_kernel<<<(16 * 32 * 256 + 255) / 256, 256, 0, stream>>>(upd_w2, upd_w2sw, 512, 512, 512);
    wswz_kernel<<<(33 * 256 + 255) / 256, 256, 0, stream>>>(em_w1, em_w1sw, 1056, 16, 1040);
    wswz_kernel<<<1, 256, 0, stream>>>(em_w2, em_w2sw, 32, 16, 16);

    float* h_out = (float*)d_out;
    float* e_out = (float*)d_out + (size_t)NN * HH;

    edge_kernel<<<EE / 16, 256, 0, stream>>>(h, edge_attr, edge_index,
        msg_w1sw, msg_b1, msg_w2sw, msg_b2, em_w1sw, em_b1, em_w2sw, em_b2,
        eg_w, eg_b, en_g, en_b, agg, e_out);

    node_kernel<<<NN / 16, 256, 0, stream>>>(h, agg, upd_w1sw, upd_b1,
        upd_w2sw, upd_b2, norm_g, norm_b, h_out);
}
